// VAMWOD_60000693125720
// MI455X (gfx1250) — matrix operand order repaired and checked by reading, not compiled, not run on hardware
//
#include <hip/hip_runtime.h>
#include <math.h>

typedef __attribute__((ext_vector_type(16))) _Float16 v16h;
typedef __attribute__((ext_vector_type(8)))  _Float16 v8h;
typedef __attribute__((ext_vector_type(4)))  _Float16 v4h;
typedef __attribute__((ext_vector_type(8)))  float    v8f;
typedef __attribute__((ext_vector_type(4)))  unsigned int u32x4;
typedef __attribute__((ext_vector_type(8)))  int      i32x8;
typedef __attribute__((ext_vector_type(4)))  int      i32x4;

#define N_NODES 16384
#define K_CTX   32
#define IN_F    608
#define HID     384
#define COLS2   768          // concat(W_i, W_j) output columns
#define ALPHA   0.2f
#define NEG_INF -9e15f

#define LDSS    40           // 32 K-halves padded to 40 -> conflict-free b128 frag reads
#define HALF_TILE (128 * LDSS)          // halves per tile
#define TILEB   (HALF_TILE * 2)         // bytes per tile (10240)

#if defined(__AMDGCN__) && __has_builtin(__builtin_amdgcn_tensor_load_to_lds) && __has_builtin(__builtin_amdgcn_s_wait_tensorcnt)
#define USE_TDM 1
#else
#define USE_TDM 0
#endif

// -------------------------------------------------------------------------
// Pack 1: h_i f32 -> f16
// -------------------------------------------------------------------------
__global__ __launch_bounds__(256)
void pack_a(const float* __restrict__ A, _Float16* __restrict__ Ah)
{
    size_t i = ((size_t)blockIdx.x * 256 + threadIdx.x) * 4;
    float4 v = *(const float4*)(A + i);
    v4h h;
    h[0] = (_Float16)v.x; h[1] = (_Float16)v.y;
    h[2] = (_Float16)v.z; h[3] = (_Float16)v.w;
    *(v4h*)(Ah + i) = h;
}

// -------------------------------------------------------------------------
// Pack 2: Wt[c][k] = f16( W[k][c] ), c in [0,768) spanning concat(W_i,W_j)
// -------------------------------------------------------------------------
__global__ __launch_bounds__(256)
void pack_w(const float* __restrict__ Wi, const float* __restrict__ Wj,
            _Float16* __restrict__ Wt)
{
    __shared__ float tile[32][33];
    const int kBase = blockIdx.x * 32;
    const int cBase = blockIdx.y * 32;
    const int tx = threadIdx.x & 31;
    const int ty = threadIdx.x >> 5;
    #pragma unroll
    for (int p = 0; p < 4; p++) {
        int k = kBase + ty + 8 * p;
        int c = cBase + tx;
        const float* src = (c < HID) ? (Wi + (size_t)k * HID + c)
                                     : (Wj + (size_t)k * HID + (c - HID));
        tile[ty + 8 * p][tx] = *src;
    }
    __syncthreads();
    #pragma unroll
    for (int p = 0; p < 4; p++) {
        int cc = ty + 8 * p;
        Wt[(size_t)(cBase + cc) * IN_F + kBase + tx] = (_Float16)tile[tx][cc];
    }
}

#if USE_TDM
// TDM 2D tile load: 32(K) x 128 rows, contiguous-K f16, row stride 608;
// D# pad (4 DWORDs per 16 DWORDs) -> 40-half LDS stride
__device__ __forceinline__ void tdm_load_tile(unsigned lds_byte_off,
                                              const _Float16* gsrc,
                                              unsigned tensor_dim1)
{
    unsigned long long ga = (unsigned long long)(uintptr_t)gsrc;
    u32x4 g0;
    g0[0] = 1u;
    g0[1] = lds_byte_off;
    g0[2] = (unsigned)ga;
    g0[3] = (unsigned)((ga >> 32) & 0x01FFFFFFull) | (2u << 30);
    i32x8 g1;
    g1[0] = (int)((1u << 16) | (1u << 20) | (3u << 22) | (3u << 25));
    g1[1] = (int)(((unsigned)IN_F & 0xFFFFu) << 16);
    g1[2] = (int)((tensor_dim1 & 0xFFFFu) << 16);
    g1[3] = (int)(((tensor_dim1 >> 16) & 0xFFFFu) | (32u << 16));
    g1[4] = 128;
    g1[5] = IN_F;
    g1[6] = 0;
    g1[7] = 0;
    i32x4 z4 = {};
#if __clang_major__ >= 23
    i32x8 z8 = {};
    __builtin_amdgcn_tensor_load_to_lds(g0, g1, z4, z4, z8, 0);
#else
    __builtin_amdgcn_tensor_load_to_lds(g0, g1, z4, z4, 0);
#endif
}
#endif

// -------------------------------------------------------------------------
// GEMM: Whh[:, 0:768] = f16(h) @ f16([W_i | W_j]); f32 accum, f16 output
// grid (128, 6); 256 thr = 8 waves (2x4); block tile 128x128; K-step 32
// -------------------------------------------------------------------------
__global__ __launch_bounds__(256)
void gat_gemm_f16_wmma(const _Float16* __restrict__ Ah,  // [16384, 608]
                       const _Float16* __restrict__ Wt,  // [768, 608]
                       _Float16* __restrict__ Whh)       // [16384, 768]
{
    __shared__ __align__(16) _Float16 lds[4 * HALF_TILE]; // 2 bufs x (A,B)

    const int tid  = threadIdx.x;
    const int lane = tid & 31;
    const int wave = tid >> 5;
    const int waveRow = wave & 1;
    const int waveCol = wave >> 1;

    const int rowBlock = blockIdx.x * 128;
    const int colTile  = blockIdx.y;           // 0..5 over 768 columns
    const int colBlock = colTile * 128;

    v8f acc[4][2];
    const v8f vzero = {};
    #pragma unroll
    for (int mt = 0; mt < 4; mt++)
        #pragma unroll
        for (int nt = 0; nt < 2; nt++)
            acc[mt][nt] = vzero;

#if USE_TDM
    const unsigned ldsBase = (unsigned)(uintptr_t)lds;
    if (wave == 0) {
        tdm_load_tile(ldsBase,         Ah + (size_t)rowBlock * IN_F, N_NODES);
        tdm_load_tile(ldsBase + TILEB, Wt + (size_t)colBlock * IN_F, COLS2);
    }
#endif

    for (int k0 = 0; k0 < IN_F; k0 += 32) {
        const int pb = (k0 >> 5) & 1;
        const _Float16* ldsA;
        const _Float16* ldsB;
#if USE_TDM
        ldsA = lds + pb * 2 * HALF_TILE;
        ldsB = ldsA + HALF_TILE;
        if (wave == 0) {
            if (k0 + 32 < IN_F) {
                const int pn = pb ^ 1;
                unsigned boff = ldsBase + (unsigned)(pn * 2 * TILEB);
                tdm_load_tile(boff,         Ah + (size_t)rowBlock * IN_F + (k0 + 32), N_NODES);
                tdm_load_tile(boff + TILEB, Wt + (size_t)colBlock * IN_F + (k0 + 32), COLS2);
                __builtin_amdgcn_s_wait_tensorcnt(2);
            } else {
                __builtin_amdgcn_s_wait_tensorcnt(0);
            }
        }
        __syncthreads();
#else
        (void)pb;
        ldsA = lds;
        ldsB = lds + HALF_TILE;
        #pragma unroll
        for (int i = 0; i < 2; i++) {
            int ch = tid + i * 256;
            int r  = ch >> 2;
            int kc = (ch & 3) * 8;
            v8h a = *(const v8h*)(Ah + (size_t)(rowBlock + r) * IN_F + k0 + kc);
            *(v8h*)(lds + r * LDSS + kc) = a;
            v8h b = *(const v8h*)(Wt + (size_t)(colBlock + r) * IN_F + k0 + kc);
            *(v8h*)(lds + HALF_TILE + r * LDSS + kc) = b;
        }
        __syncthreads();
#endif

        v16h afrag[4];
        #pragma unroll
        for (int mt = 0; mt < 4; mt++) {
            int r  = waveRow * 64 + mt * 16 + (lane & 15);
            int kb = (lane < 16) ? 0 : 8;
            v8h lo = *(const v8h*)(ldsA + r * LDSS + kb);
            v8h hi = *(const v8h*)(ldsA + r * LDSS + kb + 16);
            afrag[mt] = __builtin_shufflevector(lo, hi,
                0,1,2,3,4,5,6,7,8,9,10,11,12,13,14,15);
        }
        v16h bfrag[2];
        #pragma unroll
        for (int nt = 0; nt < 2; nt++) {
            int c  = waveCol * 32 + nt * 16 + (lane & 15);
            int kb = (lane < 16) ? 0 : 8;
            v8h lo = *(const v8h*)(ldsB + c * LDSS + kb);
            v8h hi = *(const v8h*)(ldsB + c * LDSS + kb + 16);
            bfrag[nt] = __builtin_shufflevector(lo, hi,
                0,1,2,3,4,5,6,7,8,9,10,11,12,13,14,15);
        }

        #pragma unroll
        for (int mt = 0; mt < 4; mt++)
            #pragma unroll
            for (int nt = 0; nt < 2; nt++)
                acc[mt][nt] = __builtin_amdgcn_wmma_f32_16x16x32_f16(
                    false, afrag[mt], false, bfrag[nt],
                    (short)0, acc[mt][nt], false, false);

        __syncthreads();
    }

    // epilogue: f16 store into combined [16384, 768] buffer (halves all
    // downstream read traffic; values came from f16 operands anyway)
    #pragma unroll
    for (int mt = 0; mt < 4; mt++) {
        int rbase = rowBlock + waveRow * 64 + mt * 16 + ((lane < 16) ? 0 : 8);
        #pragma unroll
        for (int nt = 0; nt < 2; nt++) {
            int c = colBlock + waveCol * 32 + nt * 16 + (lane & 15);
            #pragma unroll
            for (int v = 0; v < 8; v++)
                Whh[(size_t)(rbase + v) * COLS2 + c] = (_Float16)acc[mt][nt][v];
        }
    }
}

// -------------------------------------------------------------------------
// per-node dots from f16 Whh: s_i[n] = Wh_i[n]·a_i + b ,  t[n] = Wh_all[n]·a_j
// -------------------------------------------------------------------------
__global__ __launch_bounds__(256)
void gat_dots(const _Float16* __restrict__ Whh,
              const float* __restrict__ a_i, const float* __restrict__ a_j,
              const float* __restrict__ a_b,
              float* __restrict__ s_i, float* __restrict__ t)
{
    const int lane = threadIdx.x & 31;
    const int wave = threadIdx.x >> 5;
    const int n = blockIdx.x * 8 + wave;
    const _Float16* row = Whh + (size_t)n * COLS2;

    float s1 = 0.0f, s2 = 0.0f;
    #pragma unroll
    for (int j = 0; j < HID / 32; j++) {
        int c = lane + 32 * j;
        s1 = fmaf((float)row[c],       a_i[c], s1);
        s2 = fmaf((float)row[HID + c], a_j[c], s2);
    }
    #pragma unroll
    for (int off = 16; off >= 1; off >>= 1) {
        s1 += __shfl_xor(s1, off, 32);
        s2 += __shfl_xor(s2, off, 32);
    }
    if (lane == 0) {
        s_i[n] = s1 + a_b[0];
        t[n]   = s2;
    }
}

// -------------------------------------------------------------------------
// scores -> LeakyReLU -> mask -> softmax (K=32 == wave32) -> f16 gather-agg
// -------------------------------------------------------------------------
__global__ __launch_bounds__(256)
void gat_attn(const int* __restrict__ ctx, const _Float16* __restrict__ Whh,
              const float* __restrict__ s_i, const float* __restrict__ t,
              float* __restrict__ out)
{
    const int lane = threadIdx.x & 31;
    const int wave = threadIdx.x >> 5;
    const int n = blockIdx.x * 8 + wave;

    int idx = ctx[(size_t)n * K_CTX + lane];
    float sc = s_i[n] + ((idx >= 0) ? t[idx] : 0.0f);
    sc = (sc > 0.0f) ? sc : ALPHA * sc;
    if (idx < 0) sc = NEG_INF;

    float m = sc;
    #pragma unroll
    for (int off = 16; off >= 1; off >>= 1)
        m = fmaxf(m, __shfl_xor(m, off, 32));
    float e = __expf(sc - m);
    float s = e;
    #pragma unroll
    for (int off = 16; off >= 1; off >>= 1)
        s += __shfl_xor(s, off, 32);
    float att = e / s;

    // lane owns H-columns {lane*4 + 128*j}; f16 gather (8B per lane per chunk),
    // f32 accumulate
    float acc[12];
    #pragma unroll
    for (int i = 0; i < 12; i++) acc[i] = 0.0f;

    for (int k = 0; k < K_CTX; k++) {
        int   rk = __shfl(idx, k, 32);
        float ak = __shfl(att, k, 32);
        if (rk >= 0) {
            const _Float16* rp = Whh + (size_t)rk * COLS2 + HID;  // Wh_all half
            #pragma unroll
            for (int j = 0; j < 3; j++) {
                v4h h = *(const v4h*)(rp + lane * 4 + 128 * j);
                acc[4 * j + 0] = fmaf(ak, (float)h[0], acc[4 * j + 0]);
                acc[4 * j + 1] = fmaf(ak, (float)h[1], acc[4 * j + 1]);
                acc[4 * j + 2] = fmaf(ak, (float)h[2], acc[4 * j + 2]);
                acc[4 * j + 3] = fmaf(ak, (float)h[3], acc[4 * j + 3]);
            }
        }
    }
    float* op = out + (size_t)n * HID;
    #pragma unroll
    for (int j = 0; j < 3; j++) {
        float4 v = { acc[4 * j + 0], acc[4 * j + 1], acc[4 * j + 2], acc[4 * j + 3] };
        *(float4*)(op + lane * 4 + 128 * j) = v;
    }
}

// -------------------------------------------------------------------------
extern "C" void kernel_launch(void* const* d_in, const int* in_sizes, int n_in,
                              void* d_out, int out_size, void* d_ws, size_t ws_size,
                              hipStream_t stream) {
    const float* h_i = (const float*)d_in[0];
    const int*   ctx = (const int*)  d_in[1];
    const float* W_i = (const float*)d_in[2];
    const float* W_j = (const float*)d_in[3];
    const float* a_i = (const float*)d_in[4];
    const float* a_j = (const float*)d_in[5];
    const float* a_b = (const float*)d_in[6];
    float* out = (float*)d_out;

    float* ws  = (float*)d_ws;
    float* s_i = ws;                                   // 16384 f32
    float* t   = s_i + N_NODES;                        // 16384 f32
    _Float16* Whh = (_Float16*)(t + N_NODES);          // 16384*768 f16
    _Float16* Ah  = Whh + (size_t)N_NODES * COLS2;     // 16384*608 f16
    _Float16* Wt  = Ah + (size_t)N_NODES * IN_F;       // 768*608 f16

    pack_a<<<(N_NODES * IN_F) / (256 * 4), 256, 0, stream>>>(h_i, Ah);
    pack_w<<<dim3(IN_F / 32, COLS2 / 32), 256, 0, stream>>>(W_i, W_j, Wt);

    gat_gemm_f16_wmma<<<dim3(N_NODES / 128, 6), 256, 0, stream>>>(Ah, Wt, Whh);

    gat_dots<<<N_NODES / 8, 256, 0, stream>>>(Whh, a_i, a_j, a_b, s_i, t);

    gat_attn<<<N_NODES / 8, 256, 0, stream>>>(ctx, Whh, s_i, t, out);
}
